// QLSTM_65481071395972
// MI455X (gfx1250) — compile-verified, matrix operand order checked by reading, not run on hardware
//
#include <hip/hip_runtime.h>

typedef __attribute__((ext_vector_type(16))) __bf16 v16bf;
typedef __attribute__((ext_vector_type(8)))  float  v8f;

#define QT   512
#define QB   256
#define QDIN 128
#define QTB  (QT * QB)   // 131072 rows of X

// ---------------------------------------------------------------------------
// fast activation helpers (single-trans-op paths: v_exp_f32, v_rcp_f32)
// ---------------------------------------------------------------------------
__device__ __forceinline__ float q_rcp(float x)  { return __builtin_amdgcn_rcpf(x); }
__device__ __forceinline__ float q_sigm(float x) { return q_rcp(1.0f + __expf(-x)); }
__device__ __forceinline__ float q_tanh(float x) { return 1.0f - 2.0f * q_rcp(1.0f + __expf(2.0f * x)); }

// ---------------------------------------------------------------------------
// Phase 0: pack W[0][:128] of each gate into the WMMA B-fragment (bf16),
// and compute swh[n] = sum W[0][128:256], cnst[n] = b[0] + theta[0].
// B-fragment layout (16-bit, 32x16): lane = N column (gate), VGPR pairs hold K
// with K split [0..7]/[16..23] (lanes 0-15) and [8..15]/[24..31] (lanes 16-31)
// per 32-K chunk -- same k mapping as the A fragment, transposed roles.
// ---------------------------------------------------------------------------
__global__ void qlstm_prep(const float* Wf, const float* bf_, const float* thf,
                           const float* Wi, const float* bi_, const float* thi,
                           const float* Wu, const float* bu_, const float* thu,
                           const float* Wo, const float* bo_, const float* tho,
                           __bf16* __restrict__ bfrag, float* __restrict__ hdr) {
    const int i = threadIdx.x;                 // 512 threads
    const float* Ws[4] = {Wf, Wi, Wu, Wo};
    if (i < 512) {
        const int l  = i >> 4;                 // lane 0..31
        const int e  = i & 15;                 // element within v16bf
        const int n  = l & 15;                 // gate column
        const int kh = l >> 4;                 // lane half
        const int v  = e >> 1;                 // VGPR index 0..7
        // k within a 32-wide chunk; kernel iterates chunks, fragment reloaded
        // identically each chunk so we bake only the chunk-local mapping and
        // the GEMM adds kc*32 on the A side. Here: full K mapping for kc via
        // per-chunk reuse -> store chunk-local k, weights indexed per chunk in
        // the GEMM... we instead store 4 chunks' worth? No: B is constant per
        // chunk only if we re-pack. Simplest: bake chunk 0..3 into 4 fragments.
        const int kloc = ((v >> 2) << 4) + kh * 8 + ((v & 3) << 1) + (e & 1);
        // 4 fragments, one per 32-K chunk: bfrag[kc*512 + l*16 + e]
        #pragma unroll
        for (int kc = 0; kc < 4; ++kc) {
            const int k = kc * 32 + kloc;
            float val = (n < 4) ? Ws[n][k] : 0.0f;
            bfrag[kc * 512 + i] = (__bf16)val;
        }
    }
    if (i < 4) {
        const float* W = Ws[i];
        float s = 0.0f;
        for (int k = 128; k < 256; ++k) s += W[k];
        hdr[i] = s;
        const float* bs[4]  = {bf_, bi_, bu_, bo_};
        const float* ths[4] = {thf, thi, thu, tho};
        hdr[4 + i] = bs[i][0] + ths[i][0];
    }
}

// ---------------------------------------------------------------------------
// Phase 1: Z[n, r] = X[r, :] . Wx[n, :]   (r = t*B+b, n = gate 0..3)
// One wave per 16 rows; K=128 as 4 x v_wmma_f32_16x16x32_bf16.
// ---------------------------------------------------------------------------
__global__ void __launch_bounds__(256)
qlstm_proj(const float* __restrict__ X, const __bf16* __restrict__ bfrag,
           float* __restrict__ Z) {
    const int wave = (int)((blockIdx.x * blockDim.x + threadIdx.x) >> 5);
    const int lane = threadIdx.x & 31;
    const int m    = lane & 15;
    const int kh   = lane >> 4;
    const int row0 = wave << 4;

    const v16bf* bptr = (const v16bf*)bfrag;
    v8f acc = {};

    const float* xr = X + (size_t)(row0 + m) * QDIN + kh * 8;

    #pragma unroll
    for (int kc = 0; kc < 4; ++kc) {
        const float4* p0 = (const float4*)(xr + kc * 32);
        const float4* p1 = (const float4*)(xr + kc * 32 + 16);
        float4 f0 = p0[0], f1 = p0[1];
        float4 f2 = p1[0], f3 = p1[1];
        v16bf a;
        a[0]  = (__bf16)f0.x; a[1]  = (__bf16)f0.y; a[2]  = (__bf16)f0.z; a[3]  = (__bf16)f0.w;
        a[4]  = (__bf16)f1.x; a[5]  = (__bf16)f1.y; a[6]  = (__bf16)f1.z; a[7]  = (__bf16)f1.w;
        a[8]  = (__bf16)f2.x; a[9]  = (__bf16)f2.y; a[10] = (__bf16)f2.z; a[11] = (__bf16)f2.w;
        a[12] = (__bf16)f3.x; a[13] = (__bf16)f3.y; a[14] = (__bf16)f3.z; a[15] = (__bf16)f3.w;

        v16bf bm = bptr[kc * 32 + lane];
        acc = __builtin_amdgcn_wmma_f32_16x16x32_bf16(
            /*neg_a=*/false, a, /*neg_b=*/false, bm,
            /*c_mod=*/(short)0, acc, /*reuse_a=*/false, /*reuse_b=*/false);
    }

    // D layout: lane = N column, VGPR v = row (v + 8*half)
    const int n    = lane & 15;
    const int half = lane >> 4;
    if (n < 4) {
        #pragma unroll
        for (int v = 0; v < 8; ++v) {
            const int mr = v + (half << 3);
            Z[(size_t)n * QTB + row0 + mr] = acc[v];
        }
    }
}

// ---------------------------------------------------------------------------
// Phase 2: sequential per-batch scalar recurrence. 8 single-wave blocks.
// ---------------------------------------------------------------------------
__global__ void qlstm_scan(const float* __restrict__ Z, const float* __restrict__ hdr,
                           float* __restrict__ Hs) {
    const int b = blockIdx.x * 32 + threadIdx.x;     // 8 x 32 = 256
    const float swf = hdr[0], swi = hdr[1], swu = hdr[2], swo = hdr[3];
    const float kf  = hdr[4], ki  = hdr[5], ku  = hdr[6], ko  = hdr[7];
    const float* Zf = Z;
    const float* Zi = Z + (size_t)QTB;
    const float* Zu = Z + (size_t)2 * QTB;
    const float* Zo = Z + (size_t)3 * QTB;

    float h = 0.0f, c = 0.0f;
    for (int t = 0; t < QT; ++t) {
        const int idx = t * QB + b;
        const float af = Zf[idx] + fmaf(h, swf, kf);
        const float ai = Zi[idx] + fmaf(h, swi, ki);
        const float au = Zu[idx] + fmaf(h, swu, ku);
        const float ao = Zo[idx] + fmaf(h, swo, ko);
        const float f  = q_sigm(__cosf(af));
        const float ig = q_sigm(__cosf(ai));
        const float g  = q_tanh(__cosf(au));
        const float o  = q_sigm(__cosf(ao));
        c = fmaf(f, c, ig * g);
        h = o * q_tanh(c);
        Hs[idx] = h;
    }
    Hs[QTB + b]      = h;   // final hx scalar
    Hs[QTB + QB + b] = c;   // final cx scalar
}

// ---------------------------------------------------------------------------
// Phase 3: broadcast each scalar to 128 output floats (coalesced float4).
// d_out = outputs (T*B*128) || hx (B*128) || cx (B*128); source scalars are
// Hs[0..TB) || Hs[TB..TB+B) || Hs[TB+B..TB+2B) -- identical flattening.
// ---------------------------------------------------------------------------
__global__ void __launch_bounds__(256)
qlstm_bcast(const float* __restrict__ Hs, float4* __restrict__ out) {
    const size_t tid = (size_t)blockIdx.x * blockDim.x + threadIdx.x;
    const float s = Hs[tid >> 5];   // 32 float4 per scalar (128 floats)
    float4 r; r.x = s; r.y = s; r.z = s; r.w = s;
    out[tid] = r;
}

// ---------------------------------------------------------------------------
extern "C" void kernel_launch(void* const* d_in, const int* in_sizes, int n_in,
                              void* d_out, int out_size, void* d_ws, size_t ws_size,
                              hipStream_t stream) {
    const float* X   = (const float*)d_in[0];
    const float* Wf  = (const float*)d_in[1];
    const float* bf_ = (const float*)d_in[2];
    const float* thf = (const float*)d_in[3];
    const float* Wi  = (const float*)d_in[4];
    const float* bi_ = (const float*)d_in[5];
    const float* thi = (const float*)d_in[6];
    const float* Wu  = (const float*)d_in[7];
    const float* bu_ = (const float*)d_in[8];
    const float* thu = (const float*)d_in[9];
    const float* Wo  = (const float*)d_in[10];
    const float* bo_ = (const float*)d_in[11];
    const float* tho = (const float*)d_in[12];

    // workspace layout (bytes):
    //   [0, 4096)            : 4 x 512 bf16 B-fragments (one per K chunk)
    //   [4096, 4128)         : hdr: swh[4], cnst[4]
    //   [8192, +4*TB*4)      : Z  (gate-major projections)  = 2 MB
    //   [.., +(T+2)*B*4)     : Hs (h scalars + final h,c)   = 514 KB
    char*   ws    = (char*)d_ws;
    __bf16* bfrag = (__bf16*)ws;
    float*  hdr   = (float*)(ws + 4096);
    float*  Z     = (float*)(ws + 8192);
    float*  Hs    = (float*)(ws + 8192 + (size_t)4 * QTB * sizeof(float));

    qlstm_prep<<<1, 512, 0, stream>>>(Wf, bf_, thf, Wi, bi_, thi,
                                      Wu, bu_, thu, Wo, bo_, tho, bfrag, hdr);

    // 8192 waves (one per 16 rows) = 1024 blocks x 256 threads
    qlstm_proj<<<QTB / 16 / 8, 256, 0, stream>>>(X, bfrag, Z);

    qlstm_scan<<<8, 32, 0, stream>>>(Z, hdr, Hs);

    // (T+2)*B*128 floats / 4 per thread = (T+2)*B*32 threads
    qlstm_bcast<<<(QT + 2) * QB * 32 / 256, 256, 0, stream>>>(Hs, (float4*)d_out);
}
